// TransformerBlock_16131897164450
// MI455X (gfx1250) — matrix operand order repaired and checked by reading, not compiled, not run on hardware
//
#include <hip/hip_runtime.h>

// ---------------------------------------------------------------------------
// Transformer block for MI455X (gfx1250, wave32, WMMA + Tensor Data Mover).
// All matmuls run on v_wmma_f32_16x16x32_bf16 (f32 accumulate).
// GEMM tiles are fed by TDM (tensor_load_to_lds) with double buffering.
// ---------------------------------------------------------------------------

#define D_MODEL   1024
#define NUM_HEADS 16
#define D_HEAD    64
#define D_FF      4096
#define SEQ       2048
#define BATCH     4
#define ROWS      (BATCH * SEQ)   // 8192 token rows

typedef unsigned short u16;
typedef unsigned int   u32;
typedef unsigned long long u64;
typedef __attribute__((ext_vector_type(16))) __bf16 v16bf;
typedef __attribute__((ext_vector_type(8)))  float  v8f;
typedef __attribute__((ext_vector_type(4)))  u32    su4;
typedef __attribute__((ext_vector_type(8)))  u32    su8;

union Frag { uint4 u[2]; v16bf v; };   // 32B: one 16x32 bf16 A-fragment / 32x16 B-fragment per lane

// fp32 -> bf16 round-to-nearest-even (bit manipulation; no header dependency)
__device__ __forceinline__ u16 f2bf(float x) {
  u32 u = __float_as_uint(x);
  u32 r = (u + 0x7FFFu + ((u >> 16) & 1u)) >> 16;
  return (u16)r;
}
__device__ __forceinline__ u32 pack2(float a, float b) {
  return (u32)f2bf(a) | ((u32)f2bf(b) << 16);
}

// ---------------------------------------------------------------------------
// TDM: DMA one 128x32 bf16 tile (row stride = K elements) from global to LDS.
// LDS destination uses the padded layout: +4 DWORDs (16B) after every
// 16 DWORDs (64B = one 32-element row)  ->  40-element row stride.
//   pad_interval code 3  (8B << 3 = 64B), pad_amount code 3 (4 DWORDs)
// D# group0: count=1 | lds_addr | global_addr | type=2
// D# group1: data_size=2B, pad codes, tensor_dim0=K, tensor_dim1=rows,
//            tile_dim0=32, tile_dim1=128, dim0_stride=K
// Issued by one wave per workgroup; EXEC is ignored by TDM; completion is
// tracked with TENSORcnt and published to the group via s_barrier.
// ---------------------------------------------------------------------------
__device__ __forceinline__ void tdm_load_tile128x32(const u16* gptr, u32 ldsOff, int K) {
  const u64 ga = (u64)(uintptr_t)gptr;
  su4 g0;
  g0[0] = 1u;                                  // count=1, user descriptor
  g0[1] = ldsOff;                              // lds_addr (bytes)
  g0[2] = (u32)ga;                             // global_addr[31:0]
  g0[3] = (u32)(ga >> 32) | (2u << 30);        // global_addr[56:32] | type=2 ("image")
  const u32 td0 = (u32)K;                      // tensor_dim0 (elements)
  const u32 td1 = 128u;                        // tensor_dim1 (rows in tile window)
  su8 g1;
  g1[0] = (1u << 16) | (1u << 20) | (3u << 22) | (3u << 25); // data_size=2B, pad_enable, interval, amount
  g1[1] = (td0 & 0xFFFFu) << 16;               // tensor_dim0[15:0]  -> bits 63:48
  g1[2] = (td0 >> 16) | ((td1 & 0xFFFFu) << 16); // dim0 hi / dim1 lo
  g1[3] = (td1 >> 16) | (32u << 16);           // dim1 hi | tile_dim0 = 32
  g1[4] = 128u;                                // tile_dim1 = 128, tile_dim2 = 0
  g1[5] = (u32)K;                              // tensor_dim0_stride[31:0] (elements)
  g1[6] = 0u;                                  // stride hi | dim1_stride lo
  g1[7] = 0u;
  asm volatile("tensor_load_to_lds %0, %1" :: "s"(g0), "s"(g1) : "memory");
}

// ---------------------------------------------------------------------------
// fp32 -> bf16 elementwise convert (weights), 2 elements per thread
// ---------------------------------------------------------------------------
__global__ __launch_bounds__(256) void cvt_bf16_kernel(const float* __restrict__ in,
                                                       u16* __restrict__ out, int n) {
  int i = (blockIdx.x * 256 + threadIdx.x) * 2;
  if (i < n) {
    float2 v = *(const float2*)(in + i);
    *(u32*)(out + i) = pack2(v.x, v.y);
  }
}

// ---------------------------------------------------------------------------
// RMSNorm: one 1024-wide row per 256-thread block, bf16 output
// out = x * w / sqrt(mean(x^2) + eps)
// ---------------------------------------------------------------------------
__global__ __launch_bounds__(256) void rmsnorm_kernel(const float* __restrict__ x,
                                                      const float* __restrict__ w,
                                                      u16* __restrict__ out) {
  const int row = blockIdx.x;
  const int tid = threadIdx.x;
  const float4 xv = *(const float4*)(x + (size_t)row * D_MODEL + tid * 4);
  float ss = xv.x * xv.x + xv.y * xv.y + xv.z * xv.z + xv.w * xv.w;
#pragma unroll
  for (int m = 1; m < 32; m <<= 1) ss += __shfl_xor(ss, m, 32);
  __shared__ float red[8];
  if ((tid & 31) == 0) red[tid >> 5] = ss;
  __syncthreads();
  float tot = red[0];
#pragma unroll
  for (int i = 1; i < 8; ++i) tot += red[i];
  const float inv = rsqrtf(tot * (1.0f / D_MODEL) + 1e-5f);
  const float4 wv = *(const float4*)(w + tid * 4);
  uint2 o;
  o.x = pack2(xv.x * wv.x * inv, xv.y * wv.y * inv);
  o.y = pack2(xv.z * wv.z * inv, xv.w * wv.w * inv);
  *(uint2*)(out + (size_t)row * D_MODEL + tid * 4) = o;
}

// ---------------------------------------------------------------------------
// Tiled bf16 GEMM: C[M,N] = A[M,K] * W[N,K]^T   (i.e. X @ W.T)
// Block: 256 threads = 8 waves; block tile 128x128, K-step 32.
// Waves arranged 2(M) x 4(N); each wave computes 64x32 = 4x2 WMMA tiles.
// Tile feed: TDM double-buffered (wave 0 drives DMA, all waves run WMMA).
// LDS rows padded to 40 bf16 (80B = 20 words) -> conflict-free fragment reads.
// ---------------------------------------------------------------------------
#define GM_BF16       0   // store bf16 row-major
#define GM_GELU       1   // gelu(erf) then bf16 row-major
#define GM_ADDRES_F32 2   // f32 out = acc + residual
#define GM_VT         3   // store V transposed per-head: [B, H, dk, S] bf16

__global__ __launch_bounds__(256) void gemm_bf16_kernel(
    const u16* __restrict__ A, const u16* __restrict__ W,
    void* __restrict__ out, const float* __restrict__ res,
    int M, int N, int K, int mode) {
  __shared__ u16 lA[2][128 * 40];
  __shared__ u16 lB[2][128 * 40];
  const int tid    = threadIdx.x;
  const int lane16 = tid & 15;
  const int hi     = (tid >> 4) & 1;
  const int wave   = tid >> 5;
  const int waveM  = wave & 1;
  const int waveN  = wave >> 1;
  const int mBase  = blockIdx.y * 128;
  const int nBase  = blockIdx.x * 128;
  const int koff   = hi ? 16 : 0;   // A-fragment contiguous half (matches B layout)
  const int boff   = hi ? 16 : 0;   // B-fragment contiguous half (same k order as A)
  const bool tdmWave = (wave == 0);

  // LDS byte offsets (generic pointer low 32 bits == LDS offset per aperture map)
  const u32 offA0 = (u32)(uintptr_t)&lA[0][0];
  const u32 offA1 = (u32)(uintptr_t)&lA[1][0];
  const u32 offB0 = (u32)(uintptr_t)&lB[0][0];
  const u32 offB1 = (u32)(uintptr_t)&lB[1][0];

  const u16* Abase = A + (size_t)mBase * K;
  const u16* Wbase = W + (size_t)nBase * K;

  v8f acc[4][2] = {};

  const int nSteps = K >> 5;
  // prologue: DMA tile 0 into buffer 0
  if (tdmWave) {
    tdm_load_tile128x32(Abase, offA0, K);
    tdm_load_tile128x32(Wbase, offB0, K);
  }

  for (int s = 0; s < nSteps; ++s) {
    const int cur = s & 1;
    if (tdmWave) __builtin_amdgcn_s_wait_tensorcnt(0);  // tile s resident in LDS
    __syncthreads();                                    // publish tile s; prior compute done
    if (tdmWave && (s + 1 < nSteps)) {                  // overlap: DMA tile s+1
      const int kk = (s + 1) << 5;
      tdm_load_tile128x32(Abase + kk, cur ? offA0 : offA1, K);
      tdm_load_tile128x32(Wbase + kk, cur ? offB0 : offB1, K);
    }

    const u16* la = lA[cur];
    const u16* lb = lB[cur];
    Frag af[4], bfr[2];
#pragma unroll
    for (int mt = 0; mt < 4; ++mt) {
      const u16* p = la + (waveM * 64 + mt * 16 + lane16) * 40;
      af[mt].u[0] = *(const uint4*)(p + koff);
      af[mt].u[1] = *(const uint4*)(p + koff + 8);
    }
#pragma unroll
    for (int nt = 0; nt < 2; ++nt) {
      const u16* p = lb + (waveN * 32 + nt * 16 + lane16) * 40;
      bfr[nt].u[0] = *(const uint4*)(p + boff);
      bfr[nt].u[1] = *(const uint4*)(p + boff + 8);
    }
#pragma unroll
    for (int mt = 0; mt < 4; ++mt)
#pragma unroll
      for (int nt = 0; nt < 2; ++nt)
        acc[mt][nt] = __builtin_amdgcn_wmma_f32_16x16x32_bf16(
            false, af[mt].v, false, bfr[nt].v, (short)0, acc[mt][nt], false, false);
  }

  // ---- epilogue ----
  if (mode == GM_VT) {
    // store V transposed per head: vt[b][h][dk][s], rows contiguous in s
    u16* vt = (u16*)out;
#pragma unroll
    for (int mt = 0; mt < 4; ++mt) {
      const int m0 = mBase + waveM * 64 + mt * 16 + hi * 8;  // 8 consecutive token rows
      const int b  = m0 >> 11;                                // / SEQ
      const int s0 = m0 & (SEQ - 1);
#pragma unroll
      for (int nt = 0; nt < 2; ++nt) {
        const int n  = nBase + waveN * 32 + nt * 16 + lane16; // feature index
        const int h  = n >> 6;
        const int dk = n & 63;
        uint4 o;
        o.x = pack2(acc[mt][nt][0], acc[mt][nt][1]);
        o.y = pack2(acc[mt][nt][2], acc[mt][nt][3]);
        o.z = pack2(acc[mt][nt][4], acc[mt][nt][5]);
        o.w = pack2(acc[mt][nt][6], acc[mt][nt][7]);
        *(uint4*)(vt + ((size_t)((b * NUM_HEADS + h) * D_HEAD + dk)) * SEQ + s0) = o;
      }
    }
  } else if (mode == GM_ADDRES_F32) {
    float* o = (float*)out;
#pragma unroll
    for (int mt = 0; mt < 4; ++mt)
#pragma unroll
      for (int nt = 0; nt < 2; ++nt) {
        const int n = nBase + waveN * 32 + nt * 16 + lane16;
#pragma unroll
        for (int r = 0; r < 8; ++r) {
          const int m = mBase + waveM * 64 + mt * 16 + r + hi * 8;
          const size_t ix = (size_t)m * N + n;
          o[ix] = acc[mt][nt][r] + res[ix];
        }
      }
  } else {
    u16* o = (u16*)out;
#pragma unroll
    for (int mt = 0; mt < 4; ++mt)
#pragma unroll
      for (int nt = 0; nt < 2; ++nt) {
        const int n = nBase + waveN * 32 + nt * 16 + lane16;
#pragma unroll
        for (int r = 0; r < 8; ++r) {
          const int m = mBase + waveM * 64 + mt * 16 + r + hi * 8;
          float v = acc[mt][nt][r];
          if (mode == GM_GELU) v = 0.5f * v * (1.0f + erff(v * 0.70710678118f));
          o[(size_t)m * N + n] = f2bf(v);
        }
      }
  }
}

// ---------------------------------------------------------------------------
// Flash attention (causal), one wave per (b, h, 16-query block).
// Q,K stored [B,S,D] bf16 (head-major in row); V stored transposed [B,H,dk,S].
// QK^T and PV both run on WMMA; online softmax with per-row running max/sum
// replicated across 16-lane groups; P transposed C->A layout via wave-private LDS.
// Next key/value blocks are prefetched (global_prefetch_b8).
// ---------------------------------------------------------------------------
__global__ __launch_bounds__(256) void attn_kernel(
    const u16* __restrict__ q, const u16* __restrict__ k,
    const u16* __restrict__ vt, u16* __restrict__ outp) {
  __shared__ u16 pbuf[8 * 16 * 40];   // per-wave 16x32 P tile (rows padded to 40)
  const int tid    = threadIdx.x;
  const int lane   = tid & 31;
  const int lane16 = lane & 15;
  const int hi     = lane >> 4;
  const int wave   = tid >> 5;
  const int w      = blockIdx.x * 8 + wave;
  const int qb     = w & (SEQ / 16 - 1);   // query block in sequence
  const int bh     = w >> 7;
  const int h      = bh & (NUM_HEADS - 1);
  const int b      = bh >> 4;
  const int qbase  = qb * 16;
  const int koff   = hi ? 16 : 0;
  const int boff   = hi ? 16 : 0;
  u16* pb = pbuf + wave * 16 * 40;

  // Q block: 16 x 64 bf16 as two A-fragments (K chunks of 32)
  Frag qf[2];
  {
    const u16* qrow = q + ((size_t)(b * SEQ + qbase + lane16)) * D_MODEL + h * D_HEAD;
#pragma unroll
    for (int c = 0; c < 2; ++c) {
      qf[c].u[0] = *(const uint4*)(qrow + c * 32 + koff);
      qf[c].u[1] = *(const uint4*)(qrow + c * 32 + koff + 8);
    }
  }

  float Mr[8], Lr[8];
#pragma unroll
  for (int r = 0; r < 8; ++r) { Mr[r] = -1e30f; Lr[r] = 0.0f; }
  v8f Oacc[4] = {};   // 16 x 64 f32 accumulator (4 N-tiles)

  for (int jb = 0; jb < qbase + 16; jb += 32) {      // causal: keys <= query
    // prefetch next key/value blocks into cache while this block computes
    if (jb + 48 <= SEQ) {
      const u16* kn = k + ((size_t)(b * SEQ + jb + 32 + lane16)) * D_MODEL + h * D_HEAD;
      __builtin_prefetch(kn, 0, 1);
      const u16* vn = vt + ((size_t)((b * NUM_HEADS + h) * D_HEAD + lane16 * 4)) * SEQ + jb + 32;
      __builtin_prefetch(vn, 0, 1);
    }

    // ---- scores = Q @ K^T (two 16x16 key tiles, dk contraction = 2 x 32) ----
    v8f sc[2] = {};
#pragma unroll
    for (int t = 0; t < 2; ++t)
#pragma unroll
      for (int c = 0; c < 2; ++c) {
        Frag kf;
        const u16* krow =
            k + ((size_t)(b * SEQ + jb + t * 16 + lane16)) * D_MODEL + h * D_HEAD + c * 32;
        kf.u[0] = *(const uint4*)(krow + boff);
        kf.u[1] = *(const uint4*)(krow + boff + 8);
        sc[t] = __builtin_amdgcn_wmma_f32_16x16x32_bf16(
            false, qf[c].v, false, kf.v, (short)0, sc[t], false, false);
      }

    // ---- scale + causal mask (C layout: row = r + 8*hi, col = lane16) ----
#pragma unroll
    for (int t = 0; t < 2; ++t) {
      const int ki = jb + t * 16 + lane16;
#pragma unroll
      for (int r = 0; r < 8; ++r) {
        const int qi = qbase + r + hi * 8;
        const float v = sc[t][r] * 0.125f;          // 1/sqrt(64)
        sc[t][r] = (ki <= qi) ? v : -1e30f;
      }
    }

    // ---- online softmax: running max / sum per row (cross-lane reduce) ----
    float corr[8];
#pragma unroll
    for (int r = 0; r < 8; ++r) {
      float mx = fmaxf(sc[0][r], sc[1][r]);
#pragma unroll
      for (int m = 1; m < 16; m <<= 1) mx = fmaxf(mx, __shfl_xor(mx, m, 32));
      const float mnew = fmaxf(Mr[r], mx);
      corr[r] = __expf(Mr[r] - mnew);
      Mr[r] = mnew;
      const float p0 = __expf(sc[0][r] - mnew);
      const float p1 = __expf(sc[1][r] - mnew);
      sc[0][r] = p0; sc[1][r] = p1;
      float ps = p0 + p1;
#pragma unroll
      for (int m = 1; m < 16; m <<= 1) ps += __shfl_xor(ps, m, 32);
      Lr[r] = Lr[r] * corr[r] + ps;
    }
#pragma unroll
    for (int nt = 0; nt < 4; ++nt)
#pragma unroll
      for (int r = 0; r < 8; ++r) Oacc[nt][r] *= corr[r];

    // ---- transpose P (C layout -> A layout) through wave-private LDS ----
#pragma unroll
    for (int t = 0; t < 2; ++t)
#pragma unroll
      for (int r = 0; r < 8; ++r)
        pb[(r + hi * 8) * 40 + t * 16 + lane16] = f2bf(sc[t][r]);
    Frag pf;
    pf.u[0] = *(const uint4*)(pb + lane16 * 40 + koff);
    pf.u[1] = *(const uint4*)(pb + lane16 * 40 + koff + 8);

    // ---- O += P @ V  (V transposed layout -> contiguous B-fragment loads) ----
#pragma unroll
    for (int nt = 0; nt < 4; ++nt) {
      Frag vf;
      const u16* vrow =
          vt + ((size_t)((b * NUM_HEADS + h) * D_HEAD + nt * 16 + lane16)) * SEQ + jb;
      vf.u[0] = *(const uint4*)(vrow + boff);
      vf.u[1] = *(const uint4*)(vrow + boff + 8);
      Oacc[nt] = __builtin_amdgcn_wmma_f32_16x16x32_bf16(
          false, pf.v, false, vf.v, (short)0, Oacc[nt], false, false);
    }
  }

  // ---- finalize: divide by row sum, store bf16 [B,S,D] head-major ----
#pragma unroll
  for (int r = 0; r < 8; ++r) {
    const float inv = 1.0f / Lr[r];
    const int s = qbase + r + hi * 8;
    u16* orow = outp + ((size_t)(b * SEQ + s)) * D_MODEL + h * D_HEAD;
#pragma unroll
    for (int nt = 0; nt < 4; ++nt)
      orow[nt * 16 + lane16] = f2bf(Oacc[nt][r] * inv);
  }
}

// ---------------------------------------------------------------------------
// Host orchestration
// Inputs: 0=x 1=ln1_w 2=ln2_w 3=wq 4=wk 5=wv 6=wo 7=w1 8=w2 (all fp32)
// ---------------------------------------------------------------------------
extern "C" void kernel_launch(void* const* d_in, const int* in_sizes, int n_in,
                              void* d_out, int out_size, void* d_ws, size_t ws_size,
                              hipStream_t stream) {
  const float* x    = (const float*)d_in[0];
  const float* ln1w = (const float*)d_in[1];
  const float* ln2w = (const float*)d_in[2];
  const float* wq   = (const float*)d_in[3];
  const float* wk   = (const float*)d_in[4];
  const float* wv   = (const float*)d_in[5];
  const float* wo   = (const float*)d_in[6];
  const float* w1   = (const float*)d_in[7];
  const float* w2   = (const float*)d_in[8];
  float* outp = (float*)d_out;

  char* ws = (char*)d_ws;
  size_t off = 0;
  auto alloc = [&](size_t bytes) -> char* {
    char* p = ws + off;
    off += (bytes + 255) & ~(size_t)255;
    return p;
  };
  const size_t DD = (size_t)D_MODEL * D_MODEL;
  const size_t DF = (size_t)D_FF * D_MODEL;
  u16* wqb  = (u16*)alloc(DD * 2);
  u16* wkb  = (u16*)alloc(DD * 2);
  u16* wvb  = (u16*)alloc(DD * 2);
  u16* wob  = (u16*)alloc(DD * 2);
  u16* w1b  = (u16*)alloc(DF * 2);
  u16* w2b  = (u16*)alloc(DF * 2);
  u16* xn   = (u16*)alloc((size_t)ROWS * D_MODEL * 2);
  u16* qb   = (u16*)alloc((size_t)ROWS * D_MODEL * 2);
  u16* kb   = (u16*)alloc((size_t)ROWS * D_MODEL * 2);
  u16* vtb  = (u16*)alloc((size_t)ROWS * D_MODEL * 2);   // [B,H,dk,S]
  u16* attn = (u16*)alloc((size_t)ROWS * D_MODEL * 2);
  float* yb = (float*)alloc((size_t)ROWS * D_MODEL * 4);
  u16* ynb  = (u16*)alloc((size_t)ROWS * D_MODEL * 2);
  u16* hb   = (u16*)alloc((size_t)ROWS * D_FF * 2);

  // 1) weights -> bf16
  cvt_bf16_kernel<<<DD / 512, 256, 0, stream>>>(wq, wqb, (int)DD);
  cvt_bf16_kernel<<<DD / 512, 256, 0, stream>>>(wk, wkb, (int)DD);
  cvt_bf16_kernel<<<DD / 512, 256, 0, stream>>>(wv, wvb, (int)DD);
  cvt_bf16_kernel<<<DD / 512, 256, 0, stream>>>(wo, wob, (int)DD);
  cvt_bf16_kernel<<<DF / 512, 256, 0, stream>>>(w1, w1b, (int)DF);
  cvt_bf16_kernel<<<DF / 512, 256, 0, stream>>>(w2, w2b, (int)DF);

  // 2) xn = rmsnorm(x, ln1_w)
  rmsnorm_kernel<<<ROWS, 256, 0, stream>>>(x, ln1w, xn);

  // 3) Q, K, V projections (V stored transposed per head)
  dim3 gQKV(D_MODEL / 128, ROWS / 128);
  gemm_bf16_kernel<<<gQKV, 256, 0, stream>>>(xn, wqb, qb,  nullptr, ROWS, D_MODEL, D_MODEL, GM_BF16);
  gemm_bf16_kernel<<<gQKV, 256, 0, stream>>>(xn, wkb, kb,  nullptr, ROWS, D_MODEL, D_MODEL, GM_BF16);
  gemm_bf16_kernel<<<gQKV, 256, 0, stream>>>(xn, wvb, vtb, nullptr, ROWS, D_MODEL, D_MODEL, GM_VT);

  // 4) flash attention: B*H*(S/16) = 8192 waves, 8 per block
  attn_kernel<<<(BATCH * NUM_HEADS * (SEQ / 16)) / 8, 256, 0, stream>>>(qb, kb, vtb, attn);

  // 5) y = x + attn @ wo^T
  gemm_bf16_kernel<<<gQKV, 256, 0, stream>>>(attn, wob, yb, x, ROWS, D_MODEL, D_MODEL, GM_ADDRES_F32);

  // 6) yn = rmsnorm(y, ln2_w)
  rmsnorm_kernel<<<ROWS, 256, 0, stream>>>(yb, ln2w, ynb);

  // 7) h = gelu(yn @ w1^T)
  dim3 gF1(D_FF / 128, ROWS / 128);
  gemm_bf16_kernel<<<gF1, 256, 0, stream>>>(ynb, w1b, hb, nullptr, ROWS, D_FF, D_MODEL, GM_GELU);

  // 8) out = y + h @ w2^T
  dim3 gF2(D_MODEL / 128, ROWS / 128);
  gemm_bf16_kernel<<<gF2, 256, 0, stream>>>(hb, w2b, outp, yb, ROWS, D_MODEL, D_FF, GM_ADDRES_F32);
}
